// main_net_2920577761856
// MI455X (gfx1250) — matrix operand order repaired and checked by reading, not compiled, not run on hardware
//
#include <hip/hip_runtime.h>
#include <hip/hip_bf16.h>
#include <math.h>

// ---------------- constants from the reference ----------------
#define BATCH 16
#define NPIX  12288
#define KNN   16
#define HID   128
#define ROWS  (BATCH * NPIX)   // 196608
#define MAXW  512

typedef __attribute__((ext_vector_type(16))) __bf16 v16bf;
typedef __attribute__((ext_vector_type(8)))  __bf16 bf16x8;
typedef __attribute__((ext_vector_type(8)))  float  v8f;
typedef __attribute__((ext_vector_type(4)))  unsigned int u32x4;
typedef __attribute__((ext_vector_type(8)))  int    i32x8;
typedef __attribute__((ext_vector_type(4)))  int    i32x4;

// ================= rm normalization stats (per batch row) =================
__global__ void rm_stats_kernel(const float* __restrict__ rm,
                                float* __restrict__ mean_out,
                                float* __restrict__ rstd_out) {
    int b = blockIdx.x;
    int t = threadIdx.x;
    float s = 0.f, q = 0.f;
    for (int n = t; n < NPIX; n += 256) {
        float v = rm[(size_t)b * NPIX + n];
        s += v; q += v * v;
    }
    __shared__ float ss[256], qq[256];
    ss[t] = s; qq[t] = q;
    __syncthreads();
    for (int off = 128; off > 0; off >>= 1) {
        if (t < off) { ss[t] += ss[t + off]; qq[t] += qq[t + off]; }
        __syncthreads();
    }
    if (t == 0) {
        float mu  = ss[0] / (float)NPIX;
        float var = (qq[0] - (float)NPIX * mu * mu) / (float)(NPIX - 1); // ddof=1
        float sd  = fmaxf(sqrtf(fmaxf(var, 0.f)), 1e-6f);
        mean_out[b] = mu;
        rstd_out[b] = 1.f / sd;
    }
}

// ============ fused feature build + input_proj (5->256 relu ->128 relu) ============
__global__ void input_proj_kernel(const float* __restrict__ rm,
                                  const float* __restrict__ pos,   // [NPIX,4]
                                  const float* __restrict__ mean_in,
                                  const float* __restrict__ rstd_in,
                                  const float* __restrict__ w1,    // [5,256]
                                  const float* __restrict__ b1,    // [256]
                                  const float* __restrict__ w2,    // [256,128]
                                  const float* __restrict__ b2,    // [128]
                                  __bf16* __restrict__ h0) {       // [ROWS,128]
    long node = blockIdx.x;             // 0..ROWS-1
    int  i = (int)(node % NPIX);
    int  b = (int)(node / NPIX);
    int  t = threadIdx.x;               // 128 threads

    __shared__ float x5[5];
    __shared__ float h1[2 * HID];       // 256

    if (t == 0) x5[0] = (rm[(size_t)b * NPIX + i] - mean_in[b]) * rstd_in[b];
    if (t >= 1 && t < 5) x5[t] = pos[i * 4 + (t - 1)];
    __syncthreads();

    for (int c = t; c < 2 * HID; c += 128) {
        float acc = b1[c];
        #pragma unroll
        for (int j = 0; j < 5; j++) acc += x5[j] * w1[j * 256 + c];
        h1[c] = fmaxf(acc, 0.f);
    }
    __syncthreads();

    float acc = b2[t];
    for (int j = 0; j < 2 * HID; j++) acc += h1[j] * w2[j * HID + t];
    h0[node * HID + t] = (__bf16)fmaxf(acc, 0.f);
}

// ============ weight convert+transpose: [din,dout] f32 -> [dout,din] bf16 ============
__global__ void wconv_kernel(const float* __restrict__ w, __bf16* __restrict__ wt,
                             int din, int dout) {
    int t = blockIdx.x * blockDim.x + threadIdx.x;
    int total = din * dout;
    if (t < total) {
        int r = t / dout, c = t % dout;
        wt[(size_t)c * din + r] = (__bf16)w[t];
    }
}

// ============ KNN gather-sum aggregation: agg[b,i,:] = sum_k h[b,src[i*K+k],:] ============
__global__ void aggregate_kernel(const __bf16* __restrict__ h,
                                 const int* __restrict__ src,
                                 __bf16* __restrict__ agg,
                                 int din) {
    int cpn = din >> 3;                                  // 8-wide column groups
    long t = (long)blockIdx.x * blockDim.x + threadIdx.x;
    long total = (long)ROWS * cpn;
    if (t >= total) return;
    int  cg   = (int)(t % cpn);
    long node = t / cpn;
    int  i = (int)(node % NPIX);
    long b = node / NPIX;

    float acc[8];
    #pragma unroll
    for (int e = 0; e < 8; e++) acc[e] = 0.f;

    const int* s = src + (size_t)i * KNN;
    #pragma unroll 4
    for (int k = 0; k < KNN; k++) {
        long j = s[k];
        bf16x8 v = *(const bf16x8*)(h + ((b * NPIX + j) * din) + cg * 8);
        #pragma unroll
        for (int e = 0; e < 8; e++) acc[e] += (float)v[e];
    }
    bf16x8 o;
    #pragma unroll
    for (int e = 0; e < 8; e++) o[e] = (__bf16)acc[e];
    *(bf16x8*)(agg + node * din + cg * 8) = o;
}

// ---- WMMA operand loaders (16-bit 16x32 A layout per ISA: lane half picks K group) ----
__device__ inline v16bf load_a_tile(const __bf16* rowp, int k0) {
    // element e: K = (e/8)*16 + khalf + (e%8)
    bf16x8 lo = *(const bf16x8*)(rowp + k0);
    bf16x8 hi = *(const bf16x8*)(rowp + k0 + 16);
    v16bf a;
    #pragma unroll
    for (int e = 0; e < 8; e++) { a[e] = lo[e]; a[8 + e] = hi[e]; }
    return a;
}
__device__ inline v16bf load_b_tile(const __bf16* colp, int k0) {
    // weights pre-transposed to [dout,din]: one column's K is contiguous
    bf16x8 lo = *(const bf16x8*)(colp + k0);
    bf16x8 hi = *(const bf16x8*)(colp + k0 + 16);
    v16bf bm;
    #pragma unroll
    for (int e = 0; e < 8; e++) { bm[e] = lo[e]; bm[8 + e] = hi[e]; }
    return bm;
}

// ---- TDM: DMA a 16-row x DIN-col bf16 tile (row-contiguous) from global into LDS ----
// Builds the 2D tensor D# per CDNA5 ISA 8.3/8.4: count=1, type=2 ("image"),
// data_size=1 (2 bytes), tile_dim0=DIN, tile_dim1=16, tensor_dim0_stride=DIN.
__device__ inline void tdm_load_rows16(unsigned ldsOff, const __bf16* gptr, int din) {
    unsigned long long ga = (unsigned long long)(uintptr_t)gptr;
    u32x4 g0;
    g0[0] = 1u;                                          // count=1, user mode
    g0[1] = ldsOff;                                      // lds_addr
    g0[2] = (unsigned)(ga & 0xffffffffu);                // global_addr[31:0]
    g0[3] = (unsigned)((ga >> 32) & 0x1ffffffu) | (2u << 30); // addr[56:32] | type=2
    i32x8 g1;
    g1[0] = 0x00010000;                                  // data_size = 2 bytes
    g1[1] = (int)((unsigned)din << 16);                  // tensor_dim0[15:0] in [31:16]
    g1[2] = 0;                                           // dim0 hi, dim1 lo16 (=0)
    g1[3] = 3 | (int)((unsigned)din << 16);              // dim1 hi (3<<16 rows tot), tile_dim0=din
    g1[4] = 16;                                          // tile_dim1 = 16 rows
    g1[5] = din;                                         // tensor_dim0_stride
    g1[6] = 0;
    g1[7] = 0;
    i32x4 z4 = {0, 0, 0, 0};
#if __clang_major__ >= 23
    i32x8 z8 = {0, 0, 0, 0, 0, 0, 0, 0};
    __builtin_amdgcn_tensor_load_to_lds(g0, g1, z4, z4, z8, 0);
#else
    __builtin_amdgcn_tensor_load_to_lds(g0, g1, z4, z4, 0);
#endif
}

// ============ fused GraphConv GEMM + residual + LayerNorm (templated dims) ============
// out1 = agg@rel_w + h@root_w + rel_b ; r = relu(out1) + h@res_w + res_b ; h' = LN(r)
// block = 256 threads (8 waves), owns a 16-row M tile across full DOUT.
// A operands (h, agg) staged into LDS by the Tensor Data Mover.
template <int DIN, int DOUT>
__global__ void __launch_bounds__(256)
graphconv_gemm_ln_kernel(const __bf16* __restrict__ h,      // [ROWS,DIN]
                         const __bf16* __restrict__ agg,    // [ROWS,DIN]
                         const __bf16* __restrict__ relWt,  // [DOUT,DIN]
                         const __bf16* __restrict__ rootWt, // [DOUT,DIN]
                         const __bf16* __restrict__ resWt,  // [DOUT,DIN]
                         const float* __restrict__ rel_b,
                         const float* __restrict__ res_b,
                         const float* __restrict__ lng,
                         const float* __restrict__ lnb,
                         __bf16* __restrict__ hout) {       // [ROWS,DOUT]
    constexpr int NPER = DOUT / 128;  // N-tiles per wave: 1 / 2 / 4

    const int tid  = threadIdx.x;
    const int wave = tid >> 5;
    const int lane = tid & 31;
    const int lhi  = lane >> 4;       // lane half
    const int llo  = lane & 15;
    const int m0   = blockIdx.x * 16;

    __shared__ __bf16 sH[16 * DIN];
    __shared__ __bf16 sG[16 * DIN];
    __shared__ float  lds_r[16 * DOUT];
    __shared__ float  red_s[16][16];
    __shared__ float  red_q[16][16];
    __shared__ float  s_mu[16], s_rs[16];

    // --- TDM-stage the two 16xDIN A tiles into LDS (one DMA each) ---
    if (wave == 0 && lane == 0) {
        tdm_load_rows16((unsigned)(uintptr_t)(void*)sH, h   + (size_t)m0 * DIN, DIN);
        tdm_load_rows16((unsigned)(uintptr_t)(void*)sG, agg + (size_t)m0 * DIN, DIN);
        __builtin_amdgcn_s_wait_tensorcnt(0);
    }
    __syncthreads();

    v8f c1[NPER], c2[NPER];
    const v8f vzero = {0.f,0.f,0.f,0.f,0.f,0.f,0.f,0.f};
    #pragma unroll
    for (int ni = 0; ni < NPER; ni++) { c1[ni] = vzero; c2[ni] = vzero; }

    const __bf16* sHrow = sH + llo * DIN;
    const __bf16* sGrow = sG + llo * DIN;
    const int akoff = lhi * 8;        // A per-lane K sub-offset
    const int bkoff = lhi * 8;        // B per-lane K sub-offset

    for (int kc = 0; kc < DIN; kc += 32) {
        v16bf a_h = load_a_tile(sHrow, kc + akoff);
        v16bf a_g = load_a_tile(sGrow, kc + akoff);
        #pragma unroll
        for (int ni = 0; ni < NPER; ni++) {
            int ncol = (wave + ni * 8) * 16 + llo;
            const size_t cb = (size_t)ncol * DIN;
            v16bf b_rel  = load_b_tile(relWt  + cb, kc + bkoff);
            v16bf b_root = load_b_tile(rootWt + cb, kc + bkoff);
            v16bf b_res  = load_b_tile(resWt  + cb, kc + bkoff);
            c1[ni] = __builtin_amdgcn_wmma_f32_16x16x32_bf16(
                        false, a_g, false, b_rel, (short)0, c1[ni], false, false);
            c1[ni] = __builtin_amdgcn_wmma_f32_16x16x32_bf16(
                        false, a_h, false, b_root, (short)0, c1[ni], false, false);
            c2[ni] = __builtin_amdgcn_wmma_f32_16x16x32_bf16(
                        false, a_h, false, b_res, (short)0, c2[ni], false, false);
        }
    }

    // epilogue: bias + relu + residual add into LDS (C layout: VGPR r -> M = r + 8*lhi)
    #pragma unroll
    for (int ni = 0; ni < NPER; ni++) {
        int col = (wave + ni * 8) * 16 + llo;
        float rb = rel_b[col];
        float sb = res_b[col];
        int rbase = 8 * lhi;
        #pragma unroll
        for (int r = 0; r < 8; r++) {
            float v1 = c1[ni][r] + rb;
            float v  = fmaxf(v1, 0.f) + c2[ni][r] + sb;
            lds_r[(rbase + r) * DOUT + col] = v;
        }
    }
    __syncthreads();

    // in-block LayerNorm over DOUT (population variance, eps=1e-5)
    int row = tid >> 4, sub = tid & 15;
    float s = 0.f, q = 0.f;
    #pragma unroll
    for (int c = sub; c < DOUT; c += 16) {
        float v = lds_r[row * DOUT + c];
        s += v; q += v * v;
    }
    red_s[row][sub] = s; red_q[row][sub] = q;
    __syncthreads();
    if (sub == 0) {
        float S = 0.f, Q = 0.f;
        #pragma unroll
        for (int j = 0; j < 16; j++) { S += red_s[row][j]; Q += red_q[row][j]; }
        float mu  = S / (float)DOUT;
        float var = Q / (float)DOUT - mu * mu;
        s_mu[row] = mu;
        s_rs[row] = rsqrtf(var + 1e-5f);
    }
    __syncthreads();
    float mu = s_mu[row], rs = s_rs[row];
    #pragma unroll
    for (int c = sub; c < DOUT; c += 16) {
        float v = (lds_r[row * DOUT + c] - mu) * rs * lng[c] + lnb[c];
        hout[(size_t)(m0 + row) * DOUT + c] = (__bf16)v;
    }
}

// ============ global mean pool + head + Conv1d(1,2,1) ============
__global__ void pool_head_kernel(const __bf16* __restrict__ h,     // [ROWS,128]
                                 const float* __restrict__ head_w, // [128,6]
                                 const float* __restrict__ head_b, // [6]
                                 const float* __restrict__ conv_w, // [2]
                                 const float* __restrict__ conv_b, // [2]
                                 const float* __restrict__ out_scale,
                                 float* __restrict__ out) {        // [B,2,6]
    int b = blockIdx.x;
    int t = threadIdx.x;          // 128 threads = channels
    float s = 0.f;
    const __bf16* hb = h + (size_t)b * NPIX * HID;
    for (int n = 0; n < NPIX; n++) s += (float)hb[(size_t)n * HID + t];
    __shared__ float g[HID];
    g[t] = s / (float)NPIX;
    __syncthreads();
    if (t < 6) {
        float c = head_b[t];
        for (int j = 0; j < HID; j++) c += g[j] * head_w[j * 6 + t];
        float sc = out_scale[0];
        out[b * 12 + 0 + t] = (c * conv_w[0] + conv_b[0]) * sc;
        out[b * 12 + 6 + t] = (c * conv_w[1] + conv_b[1]) * sc;
    }
}

// ============================= host launcher =============================
extern "C" void kernel_launch(void* const* d_in, const int* in_sizes, int n_in,
                              void* d_out, int out_size, void* d_ws, size_t ws_size,
                              hipStream_t stream) {
    const float* rm      = (const float*)d_in[0];
    const float* pos_enc = (const float*)d_in[1];
    const int*   src     = (const int*)d_in[2];
    // d_in[3] = dst (implicit: repeat(arange(N), K)) — unused
    const float* w_in1 = (const float*)d_in[4];
    const float* b_in1 = (const float*)d_in[5];
    const float* w_in2 = (const float*)d_in[6];
    const float* b_in2 = (const float*)d_in[7];
    const float* rel_w[4], *rel_b[4], *root_w[4], *res_w[4], *res_b[4], *ln_g[4], *ln_b[4];
    for (int i = 0; i < 4; i++) {
        rel_w[i]  = (const float*)d_in[8 + i];
        rel_b[i]  = (const float*)d_in[12 + i];
        root_w[i] = (const float*)d_in[16 + i];
        res_w[i]  = (const float*)d_in[20 + i];
        res_b[i]  = (const float*)d_in[24 + i];
        ln_g[i]   = (const float*)d_in[28 + i];
        ln_b[i]   = (const float*)d_in[32 + i];
    }
    const float* head_w    = (const float*)d_in[36];
    const float* head_b    = (const float*)d_in[37];
    const float* conv_w    = (const float*)d_in[38];
    const float* conv_b    = (const float*)d_in[39];
    const float* out_scale = (const float*)d_in[40];
    float* out = (float*)d_out;

    const int dins[4]  = {HID, 2 * HID, 4 * HID, 2 * HID};
    const int douts[4] = {2 * HID, 4 * HID, 2 * HID, HID};

    // workspace carve-up
    char* p = (char*)d_ws;
    const size_t bigElems = (size_t)ROWS * MAXW;
    __bf16* bufX   = (__bf16*)p; p += bigElems * sizeof(__bf16);
    __bf16* bufY   = (__bf16*)p; p += bigElems * sizeof(__bf16);
    __bf16* bufAgg = (__bf16*)p; p += bigElems * sizeof(__bf16);
    float*  d_mean = (float*)p;  p += BATCH * sizeof(float);
    float*  d_rstd = (float*)p;  p += BATCH * sizeof(float);
    __bf16* wT[4][3];
    for (int l = 0; l < 4; l++)
        for (int m = 0; m < 3; m++) {
            wT[l][m] = (__bf16*)p;
            p += (size_t)dins[l] * douts[l] * sizeof(__bf16);
        }

    // 1) rm stats
    rm_stats_kernel<<<BATCH, 256, 0, stream>>>(rm, d_mean, d_rstd);

    // 2) weight convert+transpose (rel, root, res per layer)
    for (int l = 0; l < 4; l++) {
        int total = dins[l] * douts[l];
        int blocks = (total + 255) / 256;
        wconv_kernel<<<blocks, 256, 0, stream>>>(rel_w[l],  wT[l][0], dins[l], douts[l]);
        wconv_kernel<<<blocks, 256, 0, stream>>>(root_w[l], wT[l][1], dins[l], douts[l]);
        wconv_kernel<<<blocks, 256, 0, stream>>>(res_w[l],  wT[l][2], dins[l], douts[l]);
    }

    // 3) fused input projection -> h0 (bf16) into bufX
    input_proj_kernel<<<ROWS, HID, 0, stream>>>(rm, pos_enc, d_mean, d_rstd,
                                                w_in1, b_in1, w_in2, b_in2, bufX);

    // 4) four GraphConv layers, ping-pong bufX <-> bufY
    __bf16* hin  = bufX;
    __bf16* hout = bufY;
    for (int l = 0; l < 4; l++) {
        int din = dins[l];
        long aggThreads = (long)ROWS * (din >> 3);
        int  aggBlocks  = (int)((aggThreads + 255) / 256);
        aggregate_kernel<<<aggBlocks, 256, 0, stream>>>(hin, src, bufAgg, din);
        switch (l) {
        case 0:
            graphconv_gemm_ln_kernel<128, 256><<<ROWS / 16, 256, 0, stream>>>(
                hin, bufAgg, wT[l][0], wT[l][1], wT[l][2],
                rel_b[l], res_b[l], ln_g[l], ln_b[l], hout);
            break;
        case 1:
            graphconv_gemm_ln_kernel<256, 512><<<ROWS / 16, 256, 0, stream>>>(
                hin, bufAgg, wT[l][0], wT[l][1], wT[l][2],
                rel_b[l], res_b[l], ln_g[l], ln_b[l], hout);
            break;
        case 2:
            graphconv_gemm_ln_kernel<512, 256><<<ROWS / 16, 256, 0, stream>>>(
                hin, bufAgg, wT[l][0], wT[l][1], wT[l][2],
                rel_b[l], res_b[l], ln_g[l], ln_b[l], hout);
            break;
        default:
            graphconv_gemm_ln_kernel<256, 128><<<ROWS / 16, 256, 0, stream>>>(
                hin, bufAgg, wT[l][0], wT[l][1], wT[l][2],
                rel_b[l], res_b[l], ln_g[l], ln_b[l], hout);
            break;
        }
        __bf16* tmp = hin; hin = hout; hout = tmp;
    }
    // after 4 swaps the final h (width 128) lives in bufX (== hin)

    // 5) pool + head + conv
    pool_head_kernel<<<BATCH, HID, 0, stream>>>(hin, head_w, head_b,
                                                conv_w, conv_b, out_scale, out);
}
